// GINDrop_38319698215464
// MI455X (gfx1250) — matrix operand order repaired and checked by reading, not compiled, not run on hardware
//
#include <hip/hip_runtime.h>
#include <hip/hip_bf16.h>

// ---------------------------------------------------------------------------
// GIN + multi-run node-dropout, fused for MI455X (gfx1250, wave32, WMMA).
//   N=50000 nodes, E=800000 edges, D=128, H=256, R=4 runs.
// Pipeline:
//   memset(agg) -> pack_weights(bf16 B-layout) -> agg_kernel(f32 atomics)
//   -> gin_mlp: async-LDS staged tiles (ASYNCcnt double buffering),
//      WMMA bf16 16x16x32 w/ f32 accum, fused mish + run-mean + residual.
// ---------------------------------------------------------------------------

typedef __attribute__((ext_vector_type(16))) __bf16 v16bf;
typedef __attribute__((ext_vector_type(8)))  float  v8f;

#define FD    128   // feature dim D
#define FH    256   // hidden dim H
#define RUNS  4

// ---- CDNA5 async global->LDS copy (guarded; sync fallback) ----------------
#if defined(__has_builtin)
#  if __has_builtin(__builtin_amdgcn_global_load_async_to_lds_b128) && \
      __has_builtin(__builtin_amdgcn_s_wait_asynccnt)
#    define GIN_ASYNC 1
#  endif
#endif
#ifndef GIN_ASYNC
#  define GIN_ASYNC 0
#endif

#if GIN_ASYNC
typedef int gin_v4i __attribute__((vector_size(16)));          // GCC-style v4i
typedef __attribute__((address_space(1))) gin_v4i GlobV4;      // global (VA64)
typedef __attribute__((address_space(3))) gin_v4i LdsV4;       // LDS (32-bit)
#endif

__device__ __forceinline__ void async_copy_b128(const float* g, float* lds) {
#if GIN_ASYNC
  GlobV4* gp = (GlobV4*)(unsigned long long)(const void*)g;
  // flat LDS address: low 32 bits are the in-LDS byte offset (aperture rule)
  LdsV4*  lp = (LdsV4*)(unsigned int)(unsigned long long)(void*)lds;
  __builtin_amdgcn_global_load_async_to_lds_b128(gp, lp, 0, 0);
#else
  *(float4*)lds = *(const float4*)g;          // synchronous fallback
#endif
}

__device__ __forceinline__ void wait_async() {
#if GIN_ASYNC
  __builtin_amdgcn_s_wait_asynccnt(0);
#endif
}

union Frag { v16bf v; unsigned int u[8]; };

__device__ __forceinline__ unsigned short f2bf(float f) {
  unsigned int u = __float_as_uint(f);
  unsigned int r = u + 0x7FFFu + ((u >> 16) & 1u);   // round-to-nearest-even
  return (unsigned short)(r >> 16);
}

__device__ __forceinline__ unsigned int pack2bf(float a, float b) {
  return (unsigned int)f2bf(a) | ((unsigned int)f2bf(b) << 16);
}

__device__ __forceinline__ float mishf(float v) {
  float sp = (v > 15.0f) ? v : log1pf(__expf(v));    // softplus, overflow-safe
  return v * tanhf(sp);
}

// ---------------------------------------------------------------------------
// Pack W1 [D,H] and W2 [H,D] (f32 row-major) into bf16 WMMA B-fragments.
// B-fragment k order used here (32x16 K-major tile; A gathers the same), per lane:
//   n = lane%16 ;  K = (lane<16 ? 0 : 16) + 2*v + h   (v=VGPR 0..7, h=half)
// Fragment storage: [(frag*32 + lane)*16 + 2v + h] -> contiguous v16bf load.
// ---------------------------------------------------------------------------
__global__ void pack_weights(const float* __restrict__ W1,
                             const float* __restrict__ W2,
                             unsigned short* __restrict__ w1p,
                             unsigned short* __restrict__ w2p) {
  int gid  = blockIdx.x * blockDim.x + threadIdx.x;   // 0..4095
  int frag = gid >> 5;
  int lane = gid & 31;
  int lo16 = lane & 15;
  int khalf = (lane < 16) ? 0 : 16;
  if (frag < 64) {                                    // W1: 4 k-steps x 16 col tiles
    int kk = frag >> 4, nt = frag & 15;
    unsigned short* dst = w1p + ((size_t)frag * 32 + lane) * 16;
    #pragma unroll
    for (int v = 0; v < 8; ++v)
      #pragma unroll
      for (int h = 0; h < 2; ++h) {
        int k = kk * 32 + khalf + 2 * v + h;
        int n = nt * 16 + lo16;
        dst[v * 2 + h] = f2bf(W1[(size_t)k * FH + n]);
      }
  } else {                                            // W2: 8 k-steps x 8 col tiles
    int f2 = frag - 64;
    int kk = f2 >> 3, ct = f2 & 7;
    unsigned short* dst = w2p + ((size_t)f2 * 32 + lane) * 16;
    #pragma unroll
    for (int v = 0; v < 8; ++v)
      #pragma unroll
      for (int h = 0; h < 2; ++h) {
        int k = kk * 32 + khalf + 2 * v + h;
        int n = ct * 16 + lo16;
        dst[v * 2 + h] = f2bf(W2[(size_t)k * FD + n]);
      }
  }
}

// ---------------------------------------------------------------------------
// Edge scatter: agg[r][dst] += (drop[r][src] ? 0 : x[src]) for all 4 runs.
// One thread = one edge x one float4 chunk (32 chunks cover D=128).
// x[src] read once (L2-resident, 25.6MB), f32 atomics into 4 run buffers.
// ---------------------------------------------------------------------------
__global__ __launch_bounds__(256)
void agg_kernel(const float4* __restrict__ x4, const int* __restrict__ ei,
                const unsigned char* __restrict__ drop, float* __restrict__ agg,
                int Nn, int En) {
  long long g = (long long)blockIdx.x * blockDim.x + threadIdx.x;
  long long e = g >> 5;
  if (e >= En) return;
  int c   = (int)(g & 31);
  int src = ei[e];
  int dst = ei[(long long)En + e];
  float4 xv = x4[(size_t)src * 32 + c];
  #pragma unroll
  for (int r = 0; r < RUNS; ++r) {
    if (!drop[(size_t)r * Nn + src]) {
      float* a = agg + ((size_t)r * Nn + dst) * FD + c * 4;
      unsafeAtomicAdd(a + 0, xv.x);
      unsafeAtomicAdd(a + 1, xv.y);
      unsafeAtomicAdd(a + 2, xv.z);
      unsafeAtomicAdd(a + 3, xv.w);
    }
  }
}

// ---------------------------------------------------------------------------
// Fused MLP per 16-node tile. x tile staged once (reused 4 runs + residual);
// agg tile double-buffered via async global->LDS, prefetching run r+1 while
// run r computes. Runs r=0..3:
//   h = mask(x)+agg  -> LDS bf16 [16x128]
//   GEMM1 (WMMA bf16) + b1 + mish -> LDS bf16 [16x256]
//   GEMM2 (WMMA bf16) accumulated across runs (linear => mean commutes)
// Epilogue: *0.25 + b2 + x residual (from LDS).
// Block = 128 threads = 4 waves; wave w owns GEMM1 cols [w*64,w*64+63],
// GEMM2 cols [w*32,w*32+31].
// ---------------------------------------------------------------------------
__global__ __launch_bounds__(128)
void gin_mlp(const float* __restrict__ x, const unsigned char* __restrict__ drop,
             const float* __restrict__ agg, const unsigned short* __restrict__ w1p,
             const unsigned short* __restrict__ w2p, const float* __restrict__ b1,
             const float* __restrict__ b2, float* __restrict__ out, int Nn) {
  __shared__ float        xBuf[16 * 128];          // f32 x tile      (8 KB)
  __shared__ float        aggBuf[2][16 * 128];     // f32 agg tiles   (16 KB)
  __shared__ unsigned int hA[16 * 64];             // 16x128 bf16     (4 KB)
  __shared__ unsigned int h2[16 * 128];            // 16x256 bf16     (8 KB)

  const int tid   = threadIdx.x;
  const int wave  = tid >> 5;
  const int lane  = tid & 31;
  const int lo16  = lane & 15;
  const int hi8   = (lane < 16) ? 0 : 8;  // A-gather K offset == C row offset
  const int n0    = blockIdx.x * 16;

  // Per-thread staging chunk: row m, 16 cols at c*16 (4 x b128).
  const int m  = tid >> 3;
  const int c  = tid & 7;
  const int nn = n0 + m;
  const bool valid = (nn < Nn);
  const int nsafe  = valid ? nn : n0;              // clamped, masked on consume
  const int lofs   = m * 128 + c * 16;

  // Kick off x tile (whole-block lifetime) + run-0 agg tile.
  {
    const float* xg = x + (size_t)nsafe * FD + c * 16;
    const float* ag = agg + (size_t)nsafe * FD + c * 16;   // run 0
    #pragma unroll
    for (int j = 0; j < 4; ++j) {
      async_copy_b128(xg + j * 4, &xBuf[lofs + j * 4]);
      async_copy_b128(ag + j * 4, &aggBuf[0][lofs + j * 4]);
    }
  }

  v8f outacc0 = {};
  v8f outacc1 = {};

  for (int r = 0; r < RUNS; ++r) {
    wait_async();                                  // agg[r] tile (and x) ready
    if (r + 1 < RUNS) {                            // prefetch next run's agg tile
      const float* ag = agg + ((size_t)(r + 1) * Nn + nsafe) * FD + c * 16;
      float* dstb = &aggBuf[(r + 1) & 1][lofs];
      #pragma unroll
      for (int j = 0; j < 4; ++j)
        async_copy_b128(ag + j * 4, dstb + j * 4);
    }

    // ---- Phase 1: build h tile (bf16) in LDS from staged tiles -----------
    {
      bool dropped = valid ? (drop[(size_t)r * Nn + nn] != 0) : true;
      float xs = (valid && !dropped) ? 1.0f : 0.0f;
      float as = valid ? 1.0f : 0.0f;
      const float* xb = &xBuf[lofs];
      const float* ab = &aggBuf[r & 1][lofs];
      #pragma unroll
      for (int j = 0; j < 4; ++j) {
        float4 xv = *(const float4*)(xb + j * 4);
        float4 av = *(const float4*)(ab + j * 4);
        hA[m * 64 + c * 8 + j * 2 + 0] =
            pack2bf(xs * xv.x + as * av.x, xs * xv.y + as * av.y);
        hA[m * 64 + c * 8 + j * 2 + 1] =
            pack2bf(xs * xv.z + as * av.z, xs * xv.w + as * av.w);
      }
    }
    __syncthreads();

    // ---- Phase 2: GEMM1 [16x128]@[128x256] + b1 + mish -------------------
    // A layout (16x32 bf16): M=lane%16, K = kk*32 + 2*hi8 + 2v+h
    Frag a[4];
    #pragma unroll
    for (int kk = 0; kk < 4; ++kk)
      #pragma unroll
      for (int v = 0; v < 8; ++v) {
        int k = kk * 32 + 2 * hi8 + 2 * v;
        a[kk].u[v] = hA[lo16 * 64 + (k >> 1)];
      }

    #pragma unroll
    for (int nt = 0; nt < 4; ++nt) {
      int colt = wave * 4 + nt;
      v8f acc = {};
      #pragma unroll
      for (int kk = 0; kk < 4; ++kk) {
        Frag b;
        b.v = *(const v16bf*)(w1p + ((size_t)(kk * 16 + colt) * 32 + lane) * 16);
        acc = __builtin_amdgcn_wmma_f32_16x16x32_bf16(
                  false, a[kk].v, false, b.v, (short)0, acc, false, false);
      }
      int   col = colt * 16 + lo16;
      float bb  = b1[col];
      unsigned short* h2s = (unsigned short*)h2;
      #pragma unroll
      for (int i = 0; i < 8; ++i) {          // C layout: row = i + hi8, col fixed
        float mv = mishf(acc[i] + bb);
        h2s[(i + hi8) * 256 + col] = f2bf(mv);
      }
    }
    __syncthreads();

    // ---- Phase 3: GEMM2 [16x256]@[256x128], accumulate across runs -------
    #pragma unroll
    for (int kk = 0; kk < 8; ++kk) {
      Frag a2;
      #pragma unroll
      for (int v = 0; v < 8; ++v) {
        int k = kk * 32 + 2 * hi8 + 2 * v;
        a2.u[v] = h2[lo16 * 128 + (k >> 1)];
      }
      Frag bA, bB;
      bA.v = *(const v16bf*)(w2p + ((size_t)(kk * 8 + wave * 2 + 0) * 32 + lane) * 16);
      bB.v = *(const v16bf*)(w2p + ((size_t)(kk * 8 + wave * 2 + 1) * 32 + lane) * 16);
      outacc0 = __builtin_amdgcn_wmma_f32_16x16x32_bf16(
                    false, a2.v, false, bA.v, (short)0, outacc0, false, false);
      outacc1 = __builtin_amdgcn_wmma_f32_16x16x32_bf16(
                    false, a2.v, false, bB.v, (short)0, outacc1, false, false);
    }
    __syncthreads();
  }

  // ---- Epilogue: mean over runs + b2 + residual x (from LDS tile) ---------
  #pragma unroll
  for (int ct = 0; ct < 2; ++ct) {
    v8f acc = ct ? outacc1 : outacc0;
    int   col = (wave * 2 + ct) * 16 + lo16;
    float bb  = b2[col];
    #pragma unroll
    for (int i = 0; i < 8; ++i) {
      int n = n0 + i + hi8;
      if (n < Nn)
        out[(size_t)n * FD + col] = acc[i] * 0.25f + bb + xBuf[(i + hi8) * 128 + col];
    }
  }
}

// ---------------------------------------------------------------------------
extern "C" void kernel_launch(void* const* d_in, const int* in_sizes, int n_in,
                              void* d_out, int out_size, void* d_ws, size_t ws_size,
                              hipStream_t stream) {
  const float*         x    = (const float*)d_in[0];
  const int*           ei   = (const int*)d_in[1];          // [2,E] int32
  const unsigned char* drop = (const unsigned char*)d_in[2];// [R,N] bool (1B)
  const float*         W1   = (const float*)d_in[3];        // [D,H]
  const float*         b1   = (const float*)d_in[4];        // [H]
  const float*         W2   = (const float*)d_in[5];        // [H,D]
  const float*         b2   = (const float*)d_in[6];        // [D]
  float*               out  = (float*)d_out;

  int Nn = in_sizes[0] / FD;   // 50000
  int En = in_sizes[1] / 2;    // 800000

  // Workspace layout: [agg f32 R*N*D][w1p bf16 64 frags][w2p bf16 64 frags]
  size_t aggBytes = (size_t)RUNS * Nn * FD * sizeof(float);       // 102.4 MB
  float*          agg = (float*)d_ws;
  unsigned short* w1p = (unsigned short*)((char*)d_ws + aggBytes);
  unsigned short* w2p = w1p + (size_t)64 * 32 * 16;

  (void)hipMemsetAsync(d_ws, 0, aggBytes, stream);
  pack_weights<<<16, 256, 0, stream>>>(W1, W2, w1p, w2p);

  long long tE      = (long long)En * 32;
  int       blocksE = (int)((tE + 255) / 256);
  agg_kernel<<<blocksE, 256, 0, stream>>>((const float4*)x, ei, drop, agg, Nn, En);

  int tiles = (Nn + 15) / 16;
  gin_mlp<<<tiles, 128, 0, stream>>>(x, drop, agg, w1p, w2p, b1, b2, out, Nn);
}
